// MultiEmbeddingBiasTower_82471962018216
// MI455X (gfx1250) — matrix operand order repaired and checked by reading, not compiled, not run on hardware
//
#include <hip/hip_runtime.h>

typedef __attribute__((ext_vector_type(16))) _Float16 v16h;
typedef __attribute__((ext_vector_type(8)))  float    v8f;

#define F        6
#define EMB_DIM  8
#define HIDDEN   32
#define ROWS     204800          // B*T = 4096*50
#define ROWS_PER_WAVE 16
#define WAVES_PER_BLOCK 8
#define THREADS  (WAVES_PER_BLOCK * 32)
#define GRID     (ROWS / (ROWS_PER_WAVE * WAVES_PER_BLOCK))   // 1600

// One wave32 computes a 16-row tile:
//   A (16x48 f16, built from embedding gathers, K padded 48->64 across two
//   16x16x32 WMMA K-chunks) x W1 (48x32 f16) -> h (16x32 f32 in two C tiles)
//   -> elu -> dot with w2 via in-wave shuffle reduction -> 16 outputs.
//
// f16 A-fragment layout for v_wmma_f32_16x16x32_f16 (16x32, per ISA 7.12.2):
//   lanes 0-15  : row M=lane,    elems 0..7 = K0..7,  elems 8..15 = K16..23
//   lanes 16-31 : row M=lane-16, elems 0..7 = K8..15, elems 8..15 = K24..31
// Since EMB_DIM==8, each 8-elem K-group is exactly one embedding row:
//   chunk0: K0..7=emb0, K8..15=emb1, K16..23=emb2, K24..31=emb3
//   chunk1: K32..39=emb4, K40..47=emb5, K48..63=0 (padding)
// So each lane fills its A fragment from 2 (chunk0) + 1 (chunk1) row gathers.
__global__ __launch_bounds__(THREADS)
void multi_emb_tower_kernel(const int*   __restrict__ ids,
                            const float* __restrict__ e0,
                            const float* __restrict__ e1,
                            const float* __restrict__ e2,
                            const float* __restrict__ e3,
                            const float* __restrict__ e4,
                            const float* __restrict__ e5,
                            const float* __restrict__ w1,
                            const float* __restrict__ b1,
                            const float* __restrict__ w2,
                            const float* __restrict__ b2,
                            float*       __restrict__ out) {
  const int lane   = threadIdx.x & 31;
  const int wave   = threadIdx.x >> 5;
  const int rowbase = (blockIdx.x * WAVES_PER_BLOCK + wave) * ROWS_PER_WAVE;
  const bool hiHalf = lane >= 16;
  const int  l15    = lane & 15;
  const int  myrow  = rowbase + l15;

  // ---- Gather this lane's embedding rows (features 0/2/4 or 1/3/5) ----
  const int fb = hiHalf ? 1 : 0;
  const long idx0 = ids[myrow * F + fb + 0];
  const long idx1 = ids[myrow * F + fb + 2];
  const long idx2 = ids[myrow * F + fb + 4];

  const float* tab0 = hiHalf ? e1 : e0;
  const float* tab1 = hiHalf ? e3 : e2;
  const float* tab2 = hiHalf ? e5 : e4;

  const float4* p0 = (const float4*)(tab0 + idx0 * EMB_DIM);
  const float4* p1 = (const float4*)(tab1 + idx1 * EMB_DIM);
  const float4* p2 = (const float4*)(tab2 + idx2 * EMB_DIM);
  float4 g0a = p0[0], g0b = p0[1];   // emb0/emb1 row  (K-group 0 or 1)
  float4 g1a = p1[0], g1b = p1[1];   // emb2/emb3 row  (K-group 2 or 3)
  float4 g2a = p2[0], g2b = p2[1];   // emb4/emb5 row  (K-group 4 or 5)

  v16h A0, A1;
  A0[0]  = (_Float16)g0a.x; A0[1]  = (_Float16)g0a.y;
  A0[2]  = (_Float16)g0a.z; A0[3]  = (_Float16)g0a.w;
  A0[4]  = (_Float16)g0b.x; A0[5]  = (_Float16)g0b.y;
  A0[6]  = (_Float16)g0b.z; A0[7]  = (_Float16)g0b.w;
  A0[8]  = (_Float16)g1a.x; A0[9]  = (_Float16)g1a.y;
  A0[10] = (_Float16)g1a.z; A0[11] = (_Float16)g1a.w;
  A0[12] = (_Float16)g1b.x; A0[13] = (_Float16)g1b.y;
  A0[14] = (_Float16)g1b.z; A0[15] = (_Float16)g1b.w;

  A1[0]  = (_Float16)g2a.x; A1[1]  = (_Float16)g2a.y;
  A1[2]  = (_Float16)g2a.z; A1[3]  = (_Float16)g2a.w;
  A1[4]  = (_Float16)g2b.x; A1[5]  = (_Float16)g2b.y;
  A1[6]  = (_Float16)g2b.z; A1[7]  = (_Float16)g2b.w;
#pragma unroll
  for (int j = 8; j < 16; ++j) A1[j] = (_Float16)0.0f;

  // ---- B fragments from w1 (f32 -> f16 on the fly; w1 is 6KB, L1-hot) ----
  // B layout (32x16 f16): lanes 0-15 hold K=0..7,16..23, col N=lane;
  //                       lanes 16-31 hold K=8..15,24..31, col N=lane-16.
  const int kb = hiHalf ? 8 : 0;
  v16h B00, B01, B10, B11;          // [K-chunk][N-tile]
#pragma unroll
  for (int j = 0; j < 16; ++j) {
    const int k = (j < 8) ? (kb + j) : (16 + kb + (j - 8));
    B00[j] = (_Float16)w1[k * HIDDEN + l15];
    B01[j] = (_Float16)w1[k * HIDDEN + l15 + 16];
    if (j < 8) {                    // chunk1 global K = 32+kb+j valid, 48..63 = 0
      B10[j] = (_Float16)w1[(32 + kb + j) * HIDDEN + l15];
      B11[j] = (_Float16)w1[(32 + kb + j) * HIDDEN + l15 + 16];
    } else {
      B10[j] = (_Float16)0.0f;
      B11[j] = (_Float16)0.0f;
    }
  }

  // ---- GEMM1: h[16x32] = A[16x48] @ w1[48x32] via 4 WMMAs ----
  v8f c0 = {}, c1 = {};
  c0 = __builtin_amdgcn_wmma_f32_16x16x32_f16(false, A0, false, B00, (short)0, c0, false, false);
  c0 = __builtin_amdgcn_wmma_f32_16x16x32_f16(false, A1, false, B10, (short)0, c0, false, false);
  c1 = __builtin_amdgcn_wmma_f32_16x16x32_f16(false, A0, false, B01, (short)0, c1, false, false);
  c1 = __builtin_amdgcn_wmma_f32_16x16x32_f16(false, A1, false, B11, (short)0, c1, false, false);

  // ---- bias + ELU + GEMM2 (dot with w2) + cross-half reduction ----
  // C/D layout: VGPR g, lanes 0-15 -> (M=g, N=lane); lanes 16-31 -> (M=8+g, N=lane-16)
  const float bias1a = b1[l15];
  const float bias1b = b1[l15 + 16];
  const float w2a    = w2[l15];
  const float w2b    = w2[l15 + 16];
  const float bias2  = b2[0];

  float r[8];
#pragma unroll
  for (int g = 0; g < 8; ++g) {
    float h0 = c0[g] + bias1a;
    h0 = h0 > 0.0f ? h0 : (__expf(h0) - 1.0f);      // elu, alpha=1
    float h1 = c1[g] + bias1b;
    h1 = h1 > 0.0f ? h1 : (__expf(h1) - 1.0f);
    float p = h0 * w2a + h1 * w2b;
    // reduce over the 16 lanes of this half (N dimension)
    p += __shfl_xor(p, 8, 16);
    p += __shfl_xor(p, 4, 16);
    p += __shfl_xor(p, 2, 16);
    p += __shfl_xor(p, 1, 16);
    r[g] = p + bias2;
  }

  // lane 0 holds rows rowbase+0..7, lane 16 holds rows rowbase+8..15
  if (l15 == 0) {
    float4 o0 = make_float4(r[0], r[1], r[2], r[3]);
    float4 o1 = make_float4(r[4], r[5], r[6], r[7]);
    float* dst = out + rowbase + (hiHalf ? 8 : 0);
    ((float4*)dst)[0] = o0;
    ((float4*)dst)[1] = o1;
  }
}

extern "C" void kernel_launch(void* const* d_in, const int* in_sizes, int n_in,
                              void* d_out, int out_size, void* d_ws, size_t ws_size,
                              hipStream_t stream) {
  (void)in_sizes; (void)n_in; (void)out_size; (void)d_ws; (void)ws_size;
  const int*   ids = (const int*)  d_in[0];
  const float* e0  = (const float*)d_in[1];
  const float* e1  = (const float*)d_in[2];
  const float* e2  = (const float*)d_in[3];
  const float* e3  = (const float*)d_in[4];
  const float* e4  = (const float*)d_in[5];
  const float* e5  = (const float*)d_in[6];
  const float* w1  = (const float*)d_in[7];
  const float* b1  = (const float*)d_in[8];
  const float* w2  = (const float*)d_in[9];
  const float* b2  = (const float*)d_in[10];
  float* out = (float*)d_out;

  multi_emb_tower_kernel<<<GRID, THREADS, 0, stream>>>(
      ids, e0, e1, e2, e3, e4, e5, w1, b1, w2, b2, out);
}
